// GMNAggregatorPairs_62766652064050
// MI455X (gfx1250) — matrix operand order repaired and checked by reading, not compiled, not run on hardware
//
#include <hip/hip_runtime.h>

// ---------------------------------------------------------------------------
// GMN aggregator, fused for MI455X (gfx1250, wave32, WMMA bf16 16x16x32).
//   y_wf = MLP_wf(x); y_gf = sigmoid(MLP_gf(x)); h = y_gf * y_wf
//   gemb = segment_sum(h, batch);  out = MLP_mg(gemb)
// Weights are double-buffered into LDS with GLOBAL_LOAD_ASYNC_TO_LDS_B128
// (ASYNCcnt), overlapping weight fetch with WMMA compute.
// ---------------------------------------------------------------------------

typedef __attribute__((ext_vector_type(16))) __bf16 v16bf;
typedef __attribute__((ext_vector_type(4)))  __bf16 v4bf;
typedef __attribute__((ext_vector_type(8)))  float  v8f;

#define DIM     128
#define NKB     4      // K blocks of 32 (128 / 32)
#define NNT     8      // N tiles of 16 (128 / 16)
#define NNODES  500000
#define NGRAPHS 10000
#define WMAT    (DIM * DIM)   // 16384 elements per packed matrix

__device__ __forceinline__ v8f zero_v8f() { v8f z = {}; return z; }

// Load a 16x32 bf16 A-fragment from a row-major 16x128 bf16 tile in LDS.
// Lane layout (ISA 7.12.2): lane = {kh=lane>>4, row=lane&15};
// element j -> K = kb*32 + (j>>3)*16 + kh*8 + (j&7)
__device__ __forceinline__ v16bf load_a_frag(const __bf16* t, int kb, int row, int kh) {
  v16bf a;
  const __bf16* p = t + row * DIM + kb * 32 + kh * 8;
#pragma unroll
  for (int j = 0; j < 16; ++j)
    a[j] = p[(j >> 3) * 16 + (j & 7)];
  return a;
}

// h[nt] += A(16x128) * W(128x128), W pre-packed in B-fragment order:
// packed index = ((kb*NNT + nt)*32 + lane)*16 + j, where for B:
// element j -> K = kb*32 + (j>>3)*16 + (lane>>4)*8 + (j&7), N = nt*16 + (lane&15)
__device__ __forceinline__ void linear_acc(const v16bf a[NKB], const __bf16* wp,
                                           v8f h[NNT], int lane) {
#pragma unroll
  for (int nt = 0; nt < NNT; ++nt) {
    v8f acc = h[nt];
#pragma unroll
    for (int kb = 0; kb < NKB; ++kb) {
      v16bf b = *(const v16bf*)(wp + ((kb * NNT + nt) * 32 + lane) * 16);
      acc = __builtin_amdgcn_wmma_f32_16x16x32_bf16(
          false, a[kb], false, b, (short)0, acc, false, false);
    }
    h[nt] = acc;
  }
}

// ---------------------------------------------------------------------------
// Kernel 0a: zero the graph-embedding accumulator (f32, in d_ws).
__global__ void gmn_zero_kernel(float4* __restrict__ p, int n4) {
  int i = blockIdx.x * 256 + threadIdx.x;
  if (i < n4) p[i] = make_float4(0.f, 0.f, 0.f, 0.f);
}

// Kernel 0b: convert + swizzle 6 fp32 128x128 weight matrices into bf16
// B-fragment order (one 32B contiguous chunk per lane per fragment).
__global__ void gmn_pack_kernel(const float* __restrict__ w0, const float* __restrict__ w1,
                                const float* __restrict__ w2, const float* __restrict__ w3,
                                const float* __restrict__ w4, const float* __restrict__ w5,
                                __bf16* __restrict__ wpack) {
  int tid = blockIdx.x * 256 + threadIdx.x;      // 6 * 16384 threads
  int mat = tid >> 14;
  int rem = tid & (WMAT - 1);
  const float* w = mat == 0 ? w0 : mat == 1 ? w1 : mat == 2 ? w2
                 : mat == 3 ? w3 : mat == 4 ? w4 : w5;
  int j    = rem & 15;
  int lane = (rem >> 4) & 31;
  int frag = rem >> 9;          // 0..31 = kb*8 + nt
  int nt   = frag & 7;
  int kb   = frag >> 3;
  int k    = kb * 32 + (lane >> 4) * 8 + (j >> 3) * 16 + (j & 7);
  int col  = nt * 16 + (lane & 15);
  wpack[tid] = (__bf16)w[k * DIM + col];
}

// ---------------------------------------------------------------------------
// Async weight prefetch: memory -> LDS directly, 16B per lane per issue,
// tracked by ASYNCcnt. 256 threads x 8 x 16B = 32KB per matrix.
__device__ __forceinline__ void prefetch_w(__bf16* dst_lds, const __bf16* src,
                                           int tid) {
  unsigned lbase = (unsigned)(size_t)dst_lds;     // low 32 bits = LDS offset
  const char* g = (const char*)src;
#pragma unroll
  for (int i = 0; i < 8; ++i) {
    unsigned off = (unsigned)(i * 256 + tid) * 16u;
    unsigned la  = lbase + off;
    asm volatile("global_load_async_to_lds_b128 %0, %1, %2"
                 :: "v"(la), "v"(off), "s"(g) : "memory");
  }
}

__device__ __forceinline__ void wait_w() {
  asm volatile("s_wait_asynccnt 0" ::: "memory");
  __syncthreads();
}

// ---------------------------------------------------------------------------
// Kernel 1: fused per-node MLPs + sorted segment-sum (atomics into gemb).
// One wave = one 16-node tile. 8 waves/block; weights double-buffered in LDS.
__global__ __launch_bounds__(256, 1)
void gmn_node_kernel(const float* __restrict__ x, const long long* __restrict__ batch,
                     const __bf16* __restrict__ wpack,
                     const float* __restrict__ wf_b1, const float* __restrict__ wf_b2,
                     const float* __restrict__ gf_b1, const float* __restrict__ gf_b2,
                     float* __restrict__ gemb, int n_tiles) {
  __shared__ __align__(16) __bf16 wbuf[2][WMAT];      // 64 KB: weight double-buffer
  __shared__ __align__(16) __bf16 stage[8][16 * DIM]; // 32 KB: per-wave x/h staging

  const int tid  = threadIdx.x;
  const int wave = tid >> 5;
  const int lane = tid & 31;
  const int row  = lane & 15;
  const int kh   = lane >> 4;
  const int tile = blockIdx.x * 8 + wave;
  const bool valid = tile < n_tiles;
  __bf16* lds = stage[wave];

  prefetch_w(wbuf[0], wpack + 0 * WMAT, tid);         // wf_w1 in flight

  // ---- segment ids (batch is sorted int64)
  int seg = 0, seg0 = 0;
  bool clean = false;
  long long base = (long long)tile * 16;
  if (valid) {
    seg   = (int)batch[base + row];
    seg0  = __shfl(seg, 0, 32);
    clean = __all(seg == seg0) != 0;
  }

  // ---- stage x tile (16x128 contiguous f32) into LDS as bf16
  v16bf ax[NKB];
  if (valid) {
    const float4* xv = (const float4*)(x + base * DIM);
#pragma unroll
    for (int i = 0; i < 16; ++i) {
      float4 v = xv[i * 32 + lane];
      v4bf b;
      b[0] = (__bf16)v.x; b[1] = (__bf16)v.y; b[2] = (__bf16)v.z; b[3] = (__bf16)v.w;
      ((v4bf*)lds)[i * 32 + lane] = b;
    }
    asm volatile("s_wait_dscnt 0" ::: "memory");
#pragma unroll
    for (int kb = 0; kb < NKB; ++kb) ax[kb] = load_a_frag(lds, kb, row, kh);
  }

  v8f y_wf[NNT];
  v16bf ah[NKB];

  // ================= wf path =================
  wait_w();                                           // wf_w1 ready
  prefetch_w(wbuf[1], wpack + 1 * WMAT, tid);         // wf_w2 in flight
  if (valid) {
    v8f h[NNT];
#pragma unroll
    for (int nt = 0; nt < NNT; ++nt) h[nt] = zero_v8f();
    linear_acc(ax, wbuf[0], h, lane);
#pragma unroll
    for (int nt = 0; nt < NNT; ++nt) {                // bias + relu -> LDS bf16
      float bias = wf_b1[nt * 16 + row];
#pragma unroll
      for (int r = 0; r < 8; ++r) {
        float v = h[nt][r] + bias;
        lds[(r + 8 * kh) * DIM + nt * 16 + row] = (__bf16)(v > 0.f ? v : 0.f);
      }
    }
    asm volatile("s_wait_dscnt 0" ::: "memory");
#pragma unroll
    for (int kb = 0; kb < NKB; ++kb) ah[kb] = load_a_frag(lds, kb, row, kh);
  }
  wait_w();                                           // wf_w2 ready; buf0 free
  prefetch_w(wbuf[0], wpack + 2 * WMAT, tid);         // gf_w1 in flight
  if (valid) {
#pragma unroll
    for (int nt = 0; nt < NNT; ++nt) y_wf[nt] = zero_v8f();
    linear_acc(ah, wbuf[1], y_wf, lane);
  }

  // ================= gf path =================
  wait_w();                                           // gf_w1 ready; buf1 free
  prefetch_w(wbuf[1], wpack + 3 * WMAT, tid);         // gf_w2 in flight
  if (valid) {
    v8f h[NNT];
#pragma unroll
    for (int nt = 0; nt < NNT; ++nt) h[nt] = zero_v8f();
    linear_acc(ax, wbuf[0], h, lane);
#pragma unroll
    for (int nt = 0; nt < NNT; ++nt) {
      float bias = gf_b1[nt * 16 + row];
#pragma unroll
      for (int r = 0; r < 8; ++r) {
        float v = h[nt][r] + bias;
        lds[(r + 8 * kh) * DIM + nt * 16 + row] = (__bf16)(v > 0.f ? v : 0.f);
      }
    }
    asm volatile("s_wait_dscnt 0" ::: "memory");
#pragma unroll
    for (int kb = 0; kb < NKB; ++kb) ah[kb] = load_a_frag(lds, kb, row, kh);
  }
  wait_w();                                           // gf_w2 ready
  if (valid) {
    v8f yg[NNT];
#pragma unroll
    for (int nt = 0; nt < NNT; ++nt) yg[nt] = zero_v8f();
    linear_acc(ah, wbuf[1], yg, lane);

    // ---- hammard + sorted segment-sum
#pragma unroll
    for (int nt = 0; nt < NNT; ++nt) {
      int n0 = nt * 16 + row;
      float b2w = wf_b2[n0];
      float b2g = gf_b2[n0];
      float vals[8];
      float colsum = 0.f;
#pragma unroll
      for (int r = 0; r < 8; ++r) {
        float yw = y_wf[nt][r] + b2w;
        float g  = 1.0f / (1.0f + __expf(-(yg[nt][r] + b2g)));
        vals[r]  = g * yw;
        colsum  += vals[r];
      }
      if (clean) {                                    // whole tile -> one graph
        colsum += __shfl_xor(colsum, 16, 32);
        if (kh == 0)
          atomicAdd(&gemb[(long long)seg0 * DIM + n0], colsum);
      } else {                                        // rare boundary tile
#pragma unroll
        for (int r = 0; r < 8; ++r) {
          int rseg = __shfl(seg, r + 8 * kh, 32);
          atomicAdd(&gemb[(long long)rseg * DIM + n0], vals[r]);
        }
      }
    }
  }
}

// ---------------------------------------------------------------------------
// Kernel 2: final MLP on the 10000x128 graph embeddings -> d_out (f32).
__global__ __launch_bounds__(256, 1)
void gmn_graph_kernel(const float* __restrict__ gemb, const __bf16* __restrict__ wpack,
                      const float* __restrict__ mg_b1, const float* __restrict__ mg_b2,
                      float* __restrict__ out, int n_tiles) {
  __shared__ __align__(16) __bf16 wbuf[2][WMAT];
  __shared__ __align__(16) __bf16 stage[8][16 * DIM];

  const int tid  = threadIdx.x;
  const int wave = tid >> 5;
  const int lane = tid & 31;
  const int row  = lane & 15;
  const int kh   = lane >> 4;
  const int tile = blockIdx.x * 8 + wave;
  const bool valid = tile < n_tiles;
  __bf16* lds = stage[wave];

  prefetch_w(wbuf[0], wpack + 4 * WMAT, tid);         // mg_w1 in flight

  long long base = (long long)tile * 16;
  v16bf ax[NKB], ah[NKB];
  if (valid) {
    const float4* gv = (const float4*)(gemb + base * DIM);
#pragma unroll
    for (int i = 0; i < 16; ++i) {
      float4 v = gv[i * 32 + lane];
      v4bf b;
      b[0] = (__bf16)v.x; b[1] = (__bf16)v.y; b[2] = (__bf16)v.z; b[3] = (__bf16)v.w;
      ((v4bf*)lds)[i * 32 + lane] = b;
    }
    asm volatile("s_wait_dscnt 0" ::: "memory");
#pragma unroll
    for (int kb = 0; kb < NKB; ++kb) ax[kb] = load_a_frag(lds, kb, row, kh);
  }

  wait_w();                                           // mg_w1 ready
  prefetch_w(wbuf[1], wpack + 5 * WMAT, tid);         // mg_w2 in flight
  if (valid) {
    v8f h[NNT];
#pragma unroll
    for (int nt = 0; nt < NNT; ++nt) h[nt] = zero_v8f();
    linear_acc(ax, wbuf[0], h, lane);
#pragma unroll
    for (int nt = 0; nt < NNT; ++nt) {
      float bias = mg_b1[nt * 16 + row];
#pragma unroll
      for (int r = 0; r < 8; ++r) {
        float v = h[nt][r] + bias;
        lds[(r + 8 * kh) * DIM + nt * 16 + row] = (__bf16)(v > 0.f ? v : 0.f);
      }
    }
    asm volatile("s_wait_dscnt 0" ::: "memory");
#pragma unroll
    for (int kb = 0; kb < NKB; ++kb) ah[kb] = load_a_frag(lds, kb, row, kh);
  }
  wait_w();                                           // mg_w2 ready
  if (valid) {
    v8f y[NNT];
#pragma unroll
    for (int nt = 0; nt < NNT; ++nt) y[nt] = zero_v8f();
    linear_acc(ah, wbuf[1], y, lane);
#pragma unroll
    for (int nt = 0; nt < NNT; ++nt) {
      int n0 = nt * 16 + row;
      float bias = mg_b2[n0];
#pragma unroll
      for (int r = 0; r < 8; ++r)
        out[(base + r + 8 * kh) * DIM + n0] = y[nt][r] + bias;
    }
  }
}

// ---------------------------------------------------------------------------
extern "C" void kernel_launch(void* const* d_in, const int* in_sizes, int n_in,
                              void* d_out, int out_size, void* d_ws, size_t ws_size,
                              hipStream_t stream) {
  const float*     x     = (const float*)d_in[0];
  const long long* batch = (const long long*)d_in[1];
  // d_in[2] = dim (always 0)
  const float* wf_w1 = (const float*)d_in[3];
  const float* wf_b1 = (const float*)d_in[4];
  const float* wf_w2 = (const float*)d_in[5];
  const float* wf_b2 = (const float*)d_in[6];
  const float* gf_w1 = (const float*)d_in[7];
  const float* gf_b1 = (const float*)d_in[8];
  const float* gf_w2 = (const float*)d_in[9];
  const float* gf_b2 = (const float*)d_in[10];
  const float* mg_w1 = (const float*)d_in[11];
  const float* mg_b1 = (const float*)d_in[12];
  const float* mg_w2 = (const float*)d_in[13];
  const float* mg_b2 = (const float*)d_in[14];

  float*  gemb  = (float*)d_ws;                              // 10000*128 f32
  __bf16* wpack = (__bf16*)((char*)d_ws +
                            (size_t)NGRAPHS * DIM * sizeof(float)); // 6*16384 bf16

  // 0a: zero segment accumulator
  int n4 = NGRAPHS * DIM / 4;
  gmn_zero_kernel<<<(n4 + 255) / 256, 256, 0, stream>>>((float4*)gemb, n4);

  // 0b: pack 6 weight matrices into bf16 WMMA B-fragment layout
  gmn_pack_kernel<<<(6 * WMAT) / 256, 256, 0, stream>>>(
      wf_w1, wf_w2, gf_w1, gf_w2, mg_w1, mg_w2, wpack);

  // 1: fused node MLPs + segment sum
  int n_node_tiles = NNODES / 16;                            // 31250
  gmn_node_kernel<<<(n_node_tiles + 7) / 8, 256, 0, stream>>>(
      x, batch, wpack, wf_b1, wf_b2, gf_b1, gf_b2, gemb, n_node_tiles);

  // 2: final graph MLP
  int n_graph_tiles = NGRAPHS / 16;                          // 625
  gmn_graph_kernel<<<(n_graph_tiles + 7) / 8, 256, 0, stream>>>(
      gemb, wpack, mg_b1, mg_b2, (float*)d_out, n_graph_tiles);
}
